// SVDLoRASequential_1829656068759
// MI455X (gfx1250) — compile-verified, matrix operand order checked by reading, not run on hardware
//
#include <hip/hip_runtime.h>

// ---------------------------------------------------------------------------
// SVD-LoRA fused linear for MI455X (gfx1250, wave32, WMMA, async-to-LDS).
//   out = x @ W  +  ((x @ u) * s) @ vt
// Fast path: one-shot fp32 -> bf16 hi/lo split pre-pass writes x/W into
// fragment-major swizzled tiles in workspace; main GEMM is a pure pipeline of
// global_load_async_to_lds_b128 (identity copy) -> ds_load_b128 -> 3x
// v_wmma_f32_16x16x32_bf16 per fp32 MAC tile (fp32 accumulation).
// Fallback (small ws): in-kernel split (round-2 design).
// ---------------------------------------------------------------------------

typedef __bf16          v16bf __attribute__((ext_vector_type(16)));
typedef float           v8f   __attribute__((ext_vector_type(8)));
typedef unsigned int    v8ui  __attribute__((ext_vector_type(8)));

#define TOKENS 8192
#define INF    4096
#define OUTF   4096
#define RANK   16

#define BM 128
#define BN 128
#define BK 32
#define NK (INF / BK)

#define TILE_BYTES 8192          // one bf16 plane tile: 128 rows x 32 K x 2 B

// ---- fast-path LDS: double-buffered bf16 planes only (64 KB) --------------
#define FP_AH 0
#define FP_AL 8192
#define FP_BH 16384
#define FP_BL 24576
#define FP_BUFSZ 32768
#define SMEM_FAST 65536

// ---- fallback LDS partition (96 KB) ---------------------------------------
#define OFF_FA0   0
#define OFF_FA1   16384
#define OFF_FB0   32768
#define OFF_FB1   49152
#define OFF_PAH   65536
#define OFF_PAL   73728
#define OFF_PBH   81920
#define OFF_PBL   90112
#define SMEM_FALL 98304

__device__ __forceinline__ unsigned bf16_rne(float f) {
  unsigned u = __float_as_uint(f);
  return (u + 0x7FFFu + ((u >> 16) & 1u)) >> 16;   // round-to-nearest-even
}

__device__ __forceinline__ void split2(float f, unsigned& hi, unsigned& lo) {
  unsigned h = bf16_rne(f);
  float hf = __uint_as_float(h << 16);
  lo = bf16_rne(f - hf);
  hi = h;
}

// CDNA5 async global -> LDS (B128), tracked by ASYNCcnt
__device__ __forceinline__ void async_ld_b128(unsigned lds_off, const void* g) {
  unsigned long long ga = (unsigned long long)g;
  asm volatile("global_load_async_to_lds_b128 %0, %1, off"
               :: "v"(lds_off), "v"(ga) : "memory");
}
__device__ __forceinline__ void wait_async_le8() {
  asm volatile("s_wait_asynccnt 0x8" ::: "memory");
}
__device__ __forceinline__ void wait_async_0() {
  asm volatile("s_wait_asynccnt 0x0" ::: "memory");
}

// fragment-major plane offset for quad (row, k4) of a 128 x 32 tile:
//   g = (k>>3)&1 lane group, i = (k&7) + 8*(k>>4) frag index,
//   16B chunks XOR-swizzled by (row>>2)&3 for bank-uniform b128 access.
__device__ __forceinline__ int plane_off_quad(int row, int k4) {
  int g     = (k4 >> 3) & 1;
  int i0    = (k4 & 7) + ((k4 >> 4) << 3);
  int chunk = 2 * g + (i0 >> 3);
  return row * 64 + ((chunk ^ ((row >> 2) & 3)) << 4) + ((i0 & 7) << 1);
}

// ---------------------------------------------------------------------------
// Pre-pass: x -> (xh, xl) fragment-major tiles.  Tile id = mb * NK + kt.
// ---------------------------------------------------------------------------
__global__ __launch_bounds__(256)
void SVDLoRA_cvtA_kernel(const float* __restrict__ x,
                         char* __restrict__ xh, char* __restrict__ xl) {
  int q   = blockIdx.x * 256 + threadIdx.x;   // quad index
  int row = q >> 10;                          // INF/4 = 1024 quads per row
  int kq  = (q & 1023) << 2;
  float4 v = *(const float4*)(x + (size_t)row * INF + kq);
  unsigned h0, l0, h1, l1, h2, l2, h3, l3;
  split2(v.x, h0, l0); split2(v.y, h1, l1);
  split2(v.z, h2, l2); split2(v.w, h3, l3);
  int mb = row >> 7, r = row & 127;
  int kt = kq >> 5,  k4 = kq & 31;
  size_t off = (size_t)(mb * NK + kt) * TILE_BYTES + plane_off_quad(r, k4);
  *(uint2*)(xh + off) = make_uint2(h0 | (h1 << 16), h2 | (h3 << 16));
  *(uint2*)(xl + off) = make_uint2(l0 | (l1 << 16), l2 | (l3 << 16));
}

// ---------------------------------------------------------------------------
// Pre-pass: W -> (wh, wl) fragment-major tiles (transposed by column quads).
// Tile id = nb * NK + kt.
// ---------------------------------------------------------------------------
__global__ __launch_bounds__(256)
void SVDLoRA_cvtB_kernel(const float* __restrict__ w,
                         char* __restrict__ wh, char* __restrict__ wl) {
  int t  = blockIdx.x * 256 + threadIdx.x;
  int n  = t & (OUTF - 1);                    // column
  int q  = t >> 12;                           // k-quad 0..1023
  int k0 = q << 2;
  const float* col = w + (size_t)k0 * OUTF + n;
  float v0 = col[0], v1 = col[OUTF], v2 = col[2 * OUTF], v3 = col[3 * OUTF];
  unsigned h0, l0, h1, l1, h2, l2, h3, l3;
  split2(v0, h0, l0); split2(v1, h1, l1);
  split2(v2, h2, l2); split2(v3, h3, l3);
  int nb = n >> 7, r = n & 127;
  int kt = k0 >> 5, k4 = k0 & 31;
  size_t off = (size_t)(nb * NK + kt) * TILE_BYTES + plane_off_quad(r, k4);
  *(uint2*)(wh + off) = make_uint2(h0 | (h1 << 16), h2 | (h3 << 16));
  *(uint2*)(wl + off) = make_uint2(l0 | (l1 << 16), l2 | (l3 << 16));
}

// ---------------------------------------------------------------------------
// Kernel: xu[m, r] = sum_k x[m,k] * u[k,r] * s[r]   (tiny, memory-bound)
// ---------------------------------------------------------------------------
__global__ __launch_bounds__(256)
void SVDLoRA_xu_kernel(const float* __restrict__ x,
                       const float* __restrict__ u,
                       const float* __restrict__ s,
                       float* __restrict__ xu) {
  const int m   = blockIdx.x;
  const int tid = threadIdx.x;

  float acc[16];
#pragma unroll
  for (int j = 0; j < 16; ++j) acc[j] = 0.0f;

  const float* xrow = x + (size_t)m * INF;
  for (int c = tid; c < INF; c += 256) {
    float xv = xrow[c];
    const float4* up = (const float4*)(u + (size_t)c * RANK);
    float4 u0 = up[0], u1 = up[1], u2 = up[2], u3 = up[3];
    acc[0]  += xv * u0.x; acc[1]  += xv * u0.y; acc[2]  += xv * u0.z; acc[3]  += xv * u0.w;
    acc[4]  += xv * u1.x; acc[5]  += xv * u1.y; acc[6]  += xv * u1.z; acc[7]  += xv * u1.w;
    acc[8]  += xv * u2.x; acc[9]  += xv * u2.y; acc[10] += xv * u2.z; acc[11] += xv * u2.w;
    acc[12] += xv * u3.x; acc[13] += xv * u3.y; acc[14] += xv * u3.z; acc[15] += xv * u3.w;
  }

#pragma unroll
  for (int j = 0; j < 16; ++j) {
#pragma unroll
    for (int off = 16; off > 0; off >>= 1)
      acc[j] += __shfl_down(acc[j], off, 32);
  }

  __shared__ float red[8][16];
  const int lane = tid & 31;
  const int wid  = tid >> 5;
  if (lane == 0) {
#pragma unroll
    for (int j = 0; j < 16; ++j) red[wid][j] = acc[j];
  }
  __syncthreads();
  if (tid < 16) {
    float t = 0.0f;
#pragma unroll
    for (int w2 = 0; w2 < 8; ++w2) t += red[w2][tid];
    xu[(size_t)m * RANK + tid] = t * s[tid];
  }
}

// ---------------------------------------------------------------------------
// Shared epilogue helper: out = acc + xu @ vt for one wave's 4x2 tiles.
// ---------------------------------------------------------------------------
__device__ __forceinline__ void lora_epilogue(v8f acc[4][2],
                                              const float* __restrict__ vt,
                                              const float* __restrict__ xu,
                                              float* __restrict__ out,
                                              int Mb, int Nb, int wm, int wn,
                                              int mr, int hi16) {
#pragma unroll
  for (int sn = 0; sn < 2; ++sn) {
    int gn = Nb + wn * 32 + sn * 16 + mr;
    float vtc[16];
#pragma unroll
    for (int k = 0; k < 16; ++k) vtc[k] = vt[(size_t)k * OUTF + gn];
#pragma unroll
    for (int sm = 0; sm < 4; ++sm) {
#pragma unroll
      for (int r = 0; r < 8; ++r) {
        int gm = Mb + wm * 64 + sm * 16 + hi16 * 8 + r;   // C layout: VGPR r -> M=r / r+8
        const float4* xr = (const float4*)(xu + (size_t)gm * RANK);
        float4 a0 = xr[0], a1 = xr[1], a2 = xr[2], a3 = xr[3];
        float add = a0.x * vtc[0]  + a0.y * vtc[1]  + a0.z * vtc[2]  + a0.w * vtc[3]
                  + a1.x * vtc[4]  + a1.y * vtc[5]  + a1.z * vtc[6]  + a1.w * vtc[7]
                  + a2.x * vtc[8]  + a2.y * vtc[9]  + a2.z * vtc[10] + a2.w * vtc[11]
                  + a3.x * vtc[12] + a3.y * vtc[13] + a3.z * vtc[14] + a3.w * vtc[15];
        out[(size_t)gm * OUTF + gn] = acc[sm][sn][r] + add;
      }
    }
  }
}

// ---------------------------------------------------------------------------
// FAST GEMM: pre-converted bf16 planes; async DMA does all data movement.
// Hot loop per K-step: 8 async b128, 1 wait, 1 barrier, 24 ds_load_b128,
// 24 v_wmma_f32_16x16x32_bf16 -- near-zero VALU.
// ---------------------------------------------------------------------------
__global__ __launch_bounds__(256, 2)
void SVDLoRA_gemm_fast_kernel(const char* __restrict__ xh, const char* __restrict__ xl,
                              const char* __restrict__ wh, const char* __restrict__ wl,
                              const float* __restrict__ vt,
                              const float* __restrict__ xu,
                              float* __restrict__ out) {
  extern __shared__ char smem[];
  const unsigned ldsBase = (unsigned)(size_t)(void*)smem;

  const int tid  = threadIdx.x;
  const int lane = tid & 31;
  const int wid  = tid >> 5;
  const int wm   = wid & 1;
  const int wn   = wid >> 1;
  const int nbIdx = blockIdx.x;
  const int mbIdx = blockIdx.y;

  const int mr   = lane & 15;
  const int hi16 = lane >> 4;

  v8f acc[4][2] = {};

  auto issue_tile = [&](int kt, int b) {
    const char* aH = xh + (size_t)(mbIdx * NK + kt) * TILE_BYTES + tid * 16;
    const char* aL = xl + (size_t)(mbIdx * NK + kt) * TILE_BYTES + tid * 16;
    const char* bH = wh + (size_t)(nbIdx * NK + kt) * TILE_BYTES + tid * 16;
    const char* bL = wl + (size_t)(nbIdx * NK + kt) * TILE_BYTES + tid * 16;
    unsigned d = ldsBase + (unsigned)(b * FP_BUFSZ) + (unsigned)(tid * 16);
    async_ld_b128(d + FP_AH,        aH);
    async_ld_b128(d + FP_AH + 4096, aH + 4096);
    async_ld_b128(d + FP_AL,        aL);
    async_ld_b128(d + FP_AL + 4096, aL + 4096);
    async_ld_b128(d + FP_BH,        bH);
    async_ld_b128(d + FP_BH + 4096, bH + 4096);
    async_ld_b128(d + FP_BL,        bL);
    async_ld_b128(d + FP_BL + 4096, bL + 4096);
  };

  issue_tile(0, 0);

  for (int kt = 0; kt < NK; ++kt) {
    const int buf = kt & 1;
    wait_async_0();                 // tile kt resident (this wave's DMA done)
    __syncthreads();                // everyone's DMA visible; prev compute done
    if (kt + 1 < NK) issue_tile(kt + 1, buf ^ 1);   // overlaps with compute

    const char* base = smem + buf * FP_BUFSZ;

    v16bf bhv[2], blv[2];
#pragma unroll
    for (int sn = 0; sn < 2; ++sn) {
      int rowb = wn * 32 + sn * 16 + mr;
      int sw   = (rowb >> 2) & 3;
      const char* bh = base + FP_BH + rowb * 64;
      const char* bl = base + FP_BL + rowb * 64;
      uint4 h0 = *(const uint4*)(bh + (((2 * hi16)     ^ sw) << 4));
      uint4 h1 = *(const uint4*)(bh + (((2 * hi16 + 1) ^ sw) << 4));
      uint4 l0 = *(const uint4*)(bl + (((2 * hi16)     ^ sw) << 4));
      uint4 l1 = *(const uint4*)(bl + (((2 * hi16 + 1) ^ sw) << 4));
      v8ui dh, dl;
      dh[0] = h0.x; dh[1] = h0.y; dh[2] = h0.z; dh[3] = h0.w;
      dh[4] = h1.x; dh[5] = h1.y; dh[6] = h1.z; dh[7] = h1.w;
      dl[0] = l0.x; dl[1] = l0.y; dl[2] = l0.z; dl[3] = l0.w;
      dl[4] = l1.x; dl[5] = l1.y; dl[6] = l1.z; dl[7] = l1.w;
      bhv[sn] = __builtin_bit_cast(v16bf, dh);
      blv[sn] = __builtin_bit_cast(v16bf, dl);
    }

#pragma unroll
    for (int sm = 0; sm < 4; ++sm) {
      int rowa = wm * 64 + sm * 16 + mr;
      int sw   = (rowa >> 2) & 3;
      const char* ah = base + FP_AH + rowa * 64;
      const char* al = base + FP_AL + rowa * 64;
      uint4 h0 = *(const uint4*)(ah + (((2 * hi16)     ^ sw) << 4));
      uint4 h1 = *(const uint4*)(ah + (((2 * hi16 + 1) ^ sw) << 4));
      uint4 l0 = *(const uint4*)(al + (((2 * hi16)     ^ sw) << 4));
      uint4 l1 = *(const uint4*)(al + (((2 * hi16 + 1) ^ sw) << 4));
      v8ui dh, dl;
      dh[0] = h0.x; dh[1] = h0.y; dh[2] = h0.z; dh[3] = h0.w;
      dh[4] = h1.x; dh[5] = h1.y; dh[6] = h1.z; dh[7] = h1.w;
      dl[0] = l0.x; dl[1] = l0.y; dl[2] = l0.z; dl[3] = l0.w;
      dl[4] = l1.x; dl[5] = l1.y; dl[6] = l1.z; dl[7] = l1.w;
      v16bf ahv = __builtin_bit_cast(v16bf, dh);
      v16bf alv = __builtin_bit_cast(v16bf, dl);
#pragma unroll
      for (int sn = 0; sn < 2; ++sn) {
        v8f c = acc[sm][sn];
        c = __builtin_amdgcn_wmma_f32_16x16x32_bf16(false, ahv, false, bhv[sn],
                                                    (short)0, c, false, false);
        c = __builtin_amdgcn_wmma_f32_16x16x32_bf16(false, ahv, false, blv[sn],
                                                    (short)0, c, false, false);
        c = __builtin_amdgcn_wmma_f32_16x16x32_bf16(false, alv, false, bhv[sn],
                                                    (short)0, c, false, false);
        acc[sm][sn] = c;
      }
    }
  }

  lora_epilogue(acc, vt, xu, out, mbIdx * BM, nbIdx * BN, wm, wn, mr, hi16);
}

// ---------------------------------------------------------------------------
// FALLBACK GEMM (round-2 design): in-kernel split, f32 staged via async DMA.
// ---------------------------------------------------------------------------
__global__ __launch_bounds__(256, 2)
void SVDLoRA_gemm_fallback_kernel(const float* __restrict__ x,
                                  const float* __restrict__ w,
                                  const float* __restrict__ vt,
                                  const float* __restrict__ xu,
                                  float* __restrict__ out) {
  extern __shared__ char smem[];
  const unsigned ldsBase = (unsigned)(size_t)(void*)smem;

  const int tid  = threadIdx.x;
  const int lane = tid & 31;
  const int wid  = tid >> 5;
  const int wm   = wid & 1;
  const int wn   = wid >> 1;
  const int Mb   = blockIdx.y * BM;
  const int Nb   = blockIdx.x * BN;

  const int mr   = lane & 15;
  const int hi16 = lane >> 4;

  v8f acc[4][2] = {};

  auto issue_tile = [&](int kt, int b) {
    const int k0 = kt * BK;
    const unsigned offA = ldsBase + (b ? OFF_FA1 : OFF_FA0);
    const unsigned offB = ldsBase + (b ? OFF_FB1 : OFF_FB0);
#pragma unroll
    for (int it = 0; it < 4; ++it) {
      int idx = tid + it * 256;
      int r   = idx >> 3, c4 = idx & 7;
      async_ld_b128(offA + (unsigned)(r * BK + c4 * 4) * 4,
                    x + (size_t)(Mb + r) * INF + k0 + c4 * 4);
      int rb  = idx >> 5, cb4 = idx & 31;
      async_ld_b128(offB + (unsigned)(rb * BN + cb4 * 4) * 4,
                    w + (size_t)(k0 + rb) * OUTF + Nb + cb4 * 4);
    }
  };

  issue_tile(0, 0);

  for (int kt = 0; kt < NK; ++kt) {
    const int buf = kt & 1;
    if (kt + 1 < NK) {
      issue_tile(kt + 1, buf ^ 1);
      wait_async_le8();
    } else {
      wait_async_0();
    }
    __syncthreads();

    {
      const float* fA = (const float*)(smem + (buf ? OFF_FA1 : OFF_FA0));
      const float* fB = (const float*)(smem + (buf ? OFF_FB1 : OFF_FB0));
#pragma unroll
      for (int it = 0; it < 4; ++it) {
        int idx = tid + it * 256;
        {
          int r = idx >> 3, c4 = idx & 7;
          float4 v = *(const float4*)(fA + r * BK + c4 * 4);
          unsigned h0, l0, h1, l1, h2, l2, h3, l3;
          split2(v.x, h0, l0); split2(v.y, h1, l1);
          split2(v.z, h2, l2); split2(v.w, h3, l3);
          int off = plane_off_quad(r, c4 * 4);
          *(uint2*)(smem + OFF_PAH + off) = make_uint2(h0 | (h1 << 16), h2 | (h3 << 16));
          *(uint2*)(smem + OFF_PAL + off) = make_uint2(l0 | (l1 << 16), l2 | (l3 << 16));
        }
        {
          int n = idx & 127, q = idx >> 7;
          const float* col = fB + q * 4 * BN + n;
          float v0 = col[0], v1 = col[BN], v2 = col[2 * BN], v3 = col[3 * BN];
          unsigned h0, l0, h1, l1, h2, l2, h3, l3;
          split2(v0, h0, l0); split2(v1, h1, l1);
          split2(v2, h2, l2); split2(v3, h3, l3);
          int off = plane_off_quad(n, q * 4);
          *(uint2*)(smem + OFF_PBH + off) = make_uint2(h0 | (h1 << 16), h2 | (h3 << 16));
          *(uint2*)(smem + OFF_PBL + off) = make_uint2(l0 | (l1 << 16), l2 | (l3 << 16));
        }
      }
    }
    __syncthreads();

    v16bf bhv[2], blv[2];
#pragma unroll
    for (int sn = 0; sn < 2; ++sn) {
      int rowb = wn * 32 + sn * 16 + mr;
      int sw   = (rowb >> 2) & 3;
      const char* bh = smem + OFF_PBH + rowb * 64;
      const char* bl = smem + OFF_PBL + rowb * 64;
      uint4 h0 = *(const uint4*)(bh + (((2 * hi16)     ^ sw) << 4));
      uint4 h1 = *(const uint4*)(bh + (((2 * hi16 + 1) ^ sw) << 4));
      uint4 l0 = *(const uint4*)(bl + (((2 * hi16)     ^ sw) << 4));
      uint4 l1 = *(const uint4*)(bl + (((2 * hi16 + 1) ^ sw) << 4));
      v8ui dh, dl;
      dh[0] = h0.x; dh[1] = h0.y; dh[2] = h0.z; dh[3] = h0.w;
      dh[4] = h1.x; dh[5] = h1.y; dh[6] = h1.z; dh[7] = h1.w;
      dl[0] = l0.x; dl[1] = l0.y; dl[2] = l0.z; dl[3] = l0.w;
      dl[4] = l1.x; dl[5] = l1.y; dl[6] = l1.z; dl[7] = l1.w;
      bhv[sn] = __builtin_bit_cast(v16bf, dh);
      blv[sn] = __builtin_bit_cast(v16bf, dl);
    }

#pragma unroll
    for (int sm = 0; sm < 4; ++sm) {
      int rowa = wm * 64 + sm * 16 + mr;
      int sw   = (rowa >> 2) & 3;
      const char* ah = smem + OFF_PAH + rowa * 64;
      const char* al = smem + OFF_PAL + rowa * 64;
      uint4 h0 = *(const uint4*)(ah + (((2 * hi16)     ^ sw) << 4));
      uint4 h1 = *(const uint4*)(ah + (((2 * hi16 + 1) ^ sw) << 4));
      uint4 l0 = *(const uint4*)(al + (((2 * hi16)     ^ sw) << 4));
      uint4 l1 = *(const uint4*)(al + (((2 * hi16 + 1) ^ sw) << 4));
      v8ui dh, dl;
      dh[0] = h0.x; dh[1] = h0.y; dh[2] = h0.z; dh[3] = h0.w;
      dh[4] = h1.x; dh[5] = h1.y; dh[6] = h1.z; dh[7] = h1.w;
      dl[0] = l0.x; dl[1] = l0.y; dl[2] = l0.z; dl[3] = l0.w;
      dl[4] = l1.x; dl[5] = l1.y; dl[6] = l1.z; dl[7] = l1.w;
      v16bf ahv = __builtin_bit_cast(v16bf, dh);
      v16bf alv = __builtin_bit_cast(v16bf, dl);
#pragma unroll
      for (int sn = 0; sn < 2; ++sn) {
        v8f c = acc[sm][sn];
        c = __builtin_amdgcn_wmma_f32_16x16x32_bf16(false, ahv, false, bhv[sn],
                                                    (short)0, c, false, false);
        c = __builtin_amdgcn_wmma_f32_16x16x32_bf16(false, ahv, false, blv[sn],
                                                    (short)0, c, false, false);
        c = __builtin_amdgcn_wmma_f32_16x16x32_bf16(false, alv, false, bhv[sn],
                                                    (short)0, c, false, false);
        acc[sm][sn] = c;
      }
    }
  }

  lora_epilogue(acc, vt, xu, out, Mb, Nb, wm, wn, mr, hi16);
}

// ---------------------------------------------------------------------------
extern "C" void kernel_launch(void* const* d_in, const int* in_sizes, int n_in,
                              void* d_out, int out_size, void* d_ws, size_t ws_size,
                              hipStream_t stream) {
  const float* x  = (const float*)d_in[0];   // [8192, 4096]
  const float* w  = (const float*)d_in[1];   // [4096, 4096]
  const float* u  = (const float*)d_in[2];   // [4096, 16]
  const float* s  = (const float*)d_in[3];   // [16]
  const float* vt = (const float*)d_in[4];   // [16, 4096]
  float* out = (float*)d_out;                // [8192, 4096]

  char* ws = (char*)d_ws;
  const size_t szX = (size_t)TOKENS * INF * 2;   // 64 MB per x plane
  const size_t szW = (size_t)INF * OUTF * 2;     // 32 MB per W plane
  const size_t off_xu = 0;                       // 512 KB
  const size_t off_xh = 1u << 20;
  const size_t off_xl = off_xh + szX;
  const size_t off_wh = off_xl + szX;
  const size_t off_wl = off_wh + szW;
  const size_t need   = off_wl + szW;            // ~193 MB

  float* xu = (float*)(ws + off_xu);
  SVDLoRA_xu_kernel<<<TOKENS, 256, 0, stream>>>(x, u, s, xu);

  dim3 grid(OUTF / BN, TOKENS / BM);             // 32 x 64 = 2048 blocks

  if (ws_size >= need) {
    char* xh = ws + off_xh; char* xl = ws + off_xl;
    char* wh = ws + off_wh; char* wl = ws + off_wl;
    SVDLoRA_cvtA_kernel<<<(TOKENS * (INF / 4)) / 256, 256, 0, stream>>>(x, xh, xl);
    SVDLoRA_cvtB_kernel<<<(INF * (OUTF / 4)) / 256, 256, 0, stream>>>(w, wh, wl);
    SVDLoRA_gemm_fast_kernel<<<grid, dim3(256), SMEM_FAST, stream>>>(
        xh, xl, wh, wl, vt, xu, out);
  } else {
    SVDLoRA_gemm_fallback_kernel<<<grid, dim3(256), SMEM_FALL, stream>>>(
        x, w, vt, xu, out);
  }
}
